// CumulativeLinkLoss_76312978915373
// MI455X (gfx1250) — compile-verified, matrix operand order checked by reading, not run on hardware
//
#include <hip/hip_runtime.h>
#include <hip/hip_bf16.h>

#define K_CLASSES 5
#define CLL_EPS 1e-8f

typedef float    v2f  __attribute__((ext_vector_type(2)));
typedef float    v8f  __attribute__((ext_vector_type(8)));
typedef float    vf4  __attribute__((ext_vector_type(4)));
typedef int      vi4  __attribute__((ext_vector_type(4)));
typedef _Float16 v16h __attribute__((ext_vector_type(16)));

#if defined(__has_builtin)
#  if __has_builtin(__builtin_amdgcn_wmma_f32_16x16x4_f32)
#    define HAVE_WMMA_F32X4 1
#  endif
#  if __has_builtin(__builtin_amdgcn_tanhf)
#    define FAST_TANHF(x) __builtin_amdgcn_tanhf(x)
#  elif __has_builtin(__builtin_amdgcn_tanh_f32)
#    define FAST_TANHF(x) __builtin_amdgcn_tanh_f32(x)
#  endif
#  if __has_builtin(__builtin_amdgcn_rcpf)
#    define FAST_RCPF(x) __builtin_amdgcn_rcpf(x)
#  else
#    define FAST_RCPF(x) (1.0f / (x))
#  endif
#endif

// sigmoid(t): v_tanh_f32 path = mul + trans + fma, no divide.
__device__ __forceinline__ float fast_sigmoid(float t) {
#if defined(FAST_TANHF)
    return __builtin_fmaf(0.5f, FAST_TANHF(0.5f * t), 0.5f);
#else
    return FAST_RCPF(1.0f + __expf(-t));
#endif
}

// loss(x, l) = -log( sigmoid(l - x) - sigmoid((l-1) - x) + eps ), with the
// boundary sigmoids replaced by 1 (l == K-1) and 0 (l == 0).
// Fully branchless: both sigmoids computed, edges resolved by v_cndmask.
__device__ __forceinline__ float cll_loss(float x, int l) {
    float fl  = (float)l;
    float shi = fast_sigmoid(fl - x);
    float slo = fast_sigmoid(fl - 1.0f - x);
    float hi  = (l < K_CLASSES - 1) ? shi : 1.0f;
    float lo  = (l > 0)             ? slo : 0.0f;
    return -__logf(hi - lo + CLL_EPS);
}

// Pass 1: streaming loss + WMMA accumulation -> one partial per block.
__global__ void __launch_bounds__(256)
cll_partial_kernel(const float* __restrict__ logits,
                   const int*   __restrict__ labels,
                   float* __restrict__ partials, int n) {
    const int tid    = blockIdx.x * blockDim.x + threadIdx.x;
    const int nthr   = gridDim.x * blockDim.x;
    const int stride = nthr * 4;
    // Region where every wave has a full 128-element chunk (wave base is a
    // multiple of 128 since blockDim is a multiple of 32).
    const int nvec = n & ~127;

    // Trip count is identical across the wave by construction; route it
    // through readfirstlane so the compiler emits a scalar loop that never
    // touches EXEC -> EXEC is structurally all-ones at every WMMA (ISA 7.12).
    int idx = tid * 4;
    int my_iters = (idx < nvec) ? ((nvec - idx + stride - 1) / stride) : 0;
    const int iters = __builtin_amdgcn_readfirstlane(my_iters);

    v8f acc = {};                        // 16x16 f32 accumulator (8 VGPRs)
#if defined(HAVE_WMMA_F32X4)
    v2f bones; bones[0] = 1.0f; bones[1] = 1.0f;             // B = ones (4x16)
#else
    v16h hones;
    #pragma unroll
    for (int i = 0; i < 16; ++i) hones[i] = (_Float16)1.0f;  // B = ones (32x16)
#endif

    for (int it = 0; it < iters; ++it, idx += stride) {
        vf4 x  = __builtin_nontemporal_load((const vf4*)(logits + idx));
        vi4 lb = __builtin_nontemporal_load((const vi4*)(labels + idx));
        float l0 = cll_loss(x[0], lb[0]);
        float l1 = cll_loss(x[1], lb[1]);
        float l2 = cll_loss(x[2], lb[2]);
        float l3 = cll_loss(x[3], lb[3]);
#if defined(HAVE_WMMA_F32X4)
        // D = A x ones + C : C's columns accumulate A's row sums, so one WMMA
        // folds 128 per-wave f32 values (2 pre-added pairs per lane) into acc.
        v2f a; a[0] = l0 + l1; a[1] = l2 + l3;
        acc = __builtin_amdgcn_wmma_f32_16x16x4_f32(
                  false, a, false, bones, (short)0, acc, false, false);
#else
        v16h a = {};
        a[0] = (_Float16)l0; a[1] = (_Float16)l1;
        a[2] = (_Float16)l2; a[3] = (_Float16)l3;
        acc = __builtin_amdgcn_wmma_f32_16x16x32_f16(
                  false, a, false, hones, (short)0, acc, false, false);
#endif
    }

    // Ragged tail (empty when n % 128 == 0): scalar, no WMMA, divergence OK.
    float tail = 0.0f;
    for (int i = nvec + tid; i < n; i += nthr)
        tail += cll_loss(logits[i], labels[i]);

    // C/D layout: lane n holds rows 0-7 of column n, lane n+16 rows 8-15.
    float s = acc[0] + acc[1] + acc[2] + acc[3] +
              acc[4] + acc[5] + acc[6] + acc[7];
    s += __shfl_xor(s, 16, 32);
    #pragma unroll
    for (int off = 16; off > 0; off >>= 1) tail += __shfl_xor(tail, off, 32);
    s += tail;

    __shared__ float wsum[8];
    const int lane = threadIdx.x & 31;
    const int wave = threadIdx.x >> 5;
    if (lane == 0) wsum[wave] = s;
    __syncthreads();
    if (threadIdx.x == 0) {
        float b = 0.0f;
        const int nw = blockDim.x >> 5;
        for (int w = 0; w < nw; ++w) b += wsum[w];   // fixed order: deterministic
        partials[blockIdx.x] = b;
    }
}

// Pass 2: deterministic tree reduction of block partials, scale by 1/B.
__global__ void __launch_bounds__(256)
cll_finalize_kernel(const float* __restrict__ partials, int np,
                    float* __restrict__ out, float invB) {
    __shared__ float sm[256];
    float s = 0.0f;
    for (int i = threadIdx.x; i < np; i += blockDim.x) s += partials[i];
    sm[threadIdx.x] = s;
    __syncthreads();
    for (int off = 128; off > 0; off >>= 1) {
        if (threadIdx.x < (unsigned)off) sm[threadIdx.x] += sm[threadIdx.x + off];
        __syncthreads();
    }
    if (threadIdx.x == 0) out[0] = sm[0] * invB;
}

extern "C" void kernel_launch(void* const* d_in, const int* in_sizes, int n_in,
                              void* d_out, int out_size, void* d_ws, size_t ws_size,
                              hipStream_t stream) {
    const float* logits = (const float*)d_in[0];
    const int*   labels = (const int*)d_in[1];
    float*       out    = (float*)d_out;
    float*       parts  = (float*)d_ws;
    const int n = in_sizes[0];            // 8388608

    const int NT = 256;
    int nb = 2048;                        // 2048 * 256 * 4-wide = 16 elems/thread
    if ((size_t)nb * sizeof(float) > ws_size) nb = (int)(ws_size / sizeof(float));
    if (nb < 1) nb = 1;

    cll_partial_kernel<<<dim3(nb), dim3(NT), 0, stream>>>(logits, labels, parts, n);
    cll_finalize_kernel<<<dim3(1), dim3(NT), 0, stream>>>(parts, nb, out,
                                                          1.0f / (float)n);
}
